// AppearanceEncoder_68882685493527
// MI455X (gfx1250) — matrix operand order repaired and checked by reading, not compiled, not run on hardware
//
#include <hip/hip_runtime.h>
#include <cstddef>

// ---------------- problem constants (match reference setup) ----------------
static constexpr int Bn = 4;       // batch
static constexpr int Nn = 16384;   // points
static constexpr int Sn = 1024;    // NPOINT (fps samples)
static constexpr int Pn = 256;     // I1 columns
static constexpr int Kn = 64;      // I2 columns
static constexpr int Cn = 384;     // FS channels (3 * 128)

typedef __attribute__((ext_vector_type(16))) _Float16 v16h;
typedef __attribute__((ext_vector_type(8)))  _Float16 v8h;
typedef __attribute__((ext_vector_type(8)))  float    v8f;

// ---------------------------------------------------------------------------
// Kernel 1: farthest point sampling. One block per batch, 1024 threads,
// 16 points per thread held in registers (coords + running min distance).
// Sequential 1023-step loop with wave-shuffle + LDS argmax reduction.
// ---------------------------------------------------------------------------
#define FPS_T 1024
#define FPS_PPT (Nn / FPS_T)   // 16

__global__ __launch_bounds__(FPS_T) void fps_kernel(
    const float* __restrict__ xyzrgb, int* __restrict__ fps_idx)
{
  const int b = blockIdx.x;
  const int t = threadIdx.x;
  const float* base = xyzrgb + (size_t)b * Nn * 6;

  float px[FPS_PPT], py[FPS_PPT], pz[FPS_PPT], dmin[FPS_PPT];
#pragma unroll
  for (int j = 0; j < FPS_PPT; ++j) {
    const int p = j * FPS_T + t;          // strided ownership -> coalesced
    const float* pp = base + (size_t)p * 6;
    px[j] = pp[0]; py[j] = pp[1]; pz[j] = pp[2];
    dmin[j] = 1e10f;                       // reference init 1e10
  }

  __shared__ float sval[32];
  __shared__ int   sidx[32];
  __shared__ int   sbest;

  int cur = 0;
  if (t == 0) fps_idx[(size_t)b * Sn + 0] = 0;

  const int lane = t & 31;
  const int wv   = t >> 5;

  for (int i = 1; i < Sn; ++i) {
    const float* cp = base + (size_t)cur * 6;   // broadcast load (same line)
    const float cx = cp[0], cy = cp[1], cz = cp[2];

    float bv = -1.0f;
    int   bi = 0;
#pragma unroll
    for (int j = 0; j < FPS_PPT; ++j) {
      const float dx = px[j] - cx, dy = py[j] - cy, dz = pz[j] - cz;
      const float d = dx * dx + dy * dy + dz * dz;
      dmin[j] = fminf(dmin[j], d);
      if (dmin[j] > bv) { bv = dmin[j]; bi = j * FPS_T + t; }
    }
    // wave32 argmax (ties -> smallest index, matching jnp.argmax)
#pragma unroll
    for (int off = 16; off >= 1; off >>= 1) {
      const float ov = __shfl_xor(bv, off, 32);
      const int   oi = __shfl_xor(bi, off, 32);
      if (ov > bv || (ov == bv && oi < bi)) { bv = ov; bi = oi; }
    }
    if (lane == 0) { sval[wv] = bv; sidx[wv] = bi; }
    __syncthreads();
    if (wv == 0) {
      bv = sval[lane]; bi = sidx[lane];
#pragma unroll
      for (int off = 16; off >= 1; off >>= 1) {
        const float ov = __shfl_xor(bv, off, 32);
        const int   oi = __shfl_xor(bi, off, 32);
        if (ov > bv || (ov == bv && oi < bi)) { bv = ov; bi = oi; }
      }
      if (lane == 0) sbest = bi;
    }
    __syncthreads();
    cur = sbest;
    if (t == 0) fps_idx[(size_t)b * Sn + i] = cur;
  }
}

// ---------------------------------------------------------------------------
// Kernel 2: gather new_xyz = xyz[fps_idx]
// ---------------------------------------------------------------------------
__global__ void gather_new_xyz_kernel(
    const float* __restrict__ xyzrgb, const int* __restrict__ fps_idx,
    float* __restrict__ new_xyz)
{
  const int t = blockIdx.x * blockDim.x + threadIdx.x;
  if (t >= Bn * Sn) return;
  const int b = t >> 10;
  const int i = fps_idx[t];
  const float* pp = xyzrgb + ((size_t)b * Nn + i) * 6;
  new_xyz[(size_t)t * 3 + 0] = pp[0];
  new_xyz[(size_t)t * 3 + 1] = pp[1];
  new_xyz[(size_t)t * 3 + 2] = pp[2];
}

// ---------------------------------------------------------------------------
// Kernel 3: ball query. One wave32 per (b,s,scale). Ordered compaction with
// ballot/popcount reproduces "first ns indices (ascending) within radius",
// padding the remainder with the first hit (reference semantics).
// ---------------------------------------------------------------------------
__global__ __launch_bounds__(128) void ball_query_kernel(
    const float* __restrict__ xyzrgb, const float* __restrict__ new_xyz,
    int* __restrict__ idx0, int* __restrict__ idx1, int* __restrict__ idx2)
{
  const int scale = blockIdx.y;
  float r2; int ns; int* out;
  if (scale == 0)      { r2 = 0.01f; ns = 32;  out = idx0; }
  else if (scale == 1) { r2 = 0.04f; ns = 64;  out = idx1; }
  else                 { r2 = 0.16f; ns = 128; out = idx2; }

  const int wid  = blockIdx.x * (blockDim.x >> 5) + (threadIdx.x >> 5);
  const int lane = threadIdx.x & 31;
  if (wid >= Bn * Sn) return;

  const int b = wid >> 10;
  const float* q = new_xyz + (size_t)wid * 3;
  const float qx = q[0], qy = q[1], qz = q[2];
  const float* base = xyzrgb + (size_t)b * Nn * 6;
  int* o = out + (size_t)wid * ns;

  int cnt = 0, first = 0;
  for (int bp = 0; bp < Nn && cnt < ns; bp += 32) {
    const int p = bp + lane;
    const float* pp = base + (size_t)p * 6;
    __builtin_prefetch(pp + 32 * 6, 0, 0);          // global_prefetch_b8
    const float dx = pp[0] - qx, dy = pp[1] - qy, dz = pp[2] - qz;
    const float d2 = dx * dx + dy * dy + dz * dz;
    const bool pred = d2 < r2;
    const unsigned m = (unsigned)__ballot(pred);    // wave32: low 32 bits
    if (cnt == 0 && m) first = bp + __ffs(m) - 1;
    const int prefix = __popc(m & ((1u << lane) - 1u));
    const int slot = cnt + prefix;
    if (pred && slot < ns) o[slot] = p;
    cnt += __popc(m);
  }
  if (cnt < ns)
    for (int sl = cnt + lane; sl < ns; sl += 32) o[sl] = first;
}

// ---------------------------------------------------------------------------
// Kernel 4: repack weights fp32 -> f16, K padded to multiple of 32, layout
// Wh[cout][Kp] so each WMMA B-fragment column is a contiguous 32B load.
// ---------------------------------------------------------------------------
__global__ void convert_weights_kernel(
    const float* __restrict__ W, const float* __restrict__ bias,
    _Float16* __restrict__ Wh, float* __restrict__ bh,
    int cout, int cin, int kp)
{
  const int t = blockIdx.x * blockDim.x + threadIdx.x;
  if (t < cout * kp) {
    const int o = t / kp, k = t % kp;
    Wh[t] = (k < cin) ? (_Float16)W[o * cin + k] : (_Float16)0.0f;
  }
  if (t < cout) bh[t] = bias[t];
}

// ---------------------------------------------------------------------------
// WMMA GEMM layer: D = relu(A(NSxKp) * W^T + b), A in LDS, W f16 in global,
// output written to LDS in f16. Fragment layouts used here (contiguous k order, same in A and B):
//   A 16x32 f16 : lane = {g=lane>>4, m=lane&15}; a[0..15]=K[g*16+..],
//                 row M = m.
//   B 32x16 f16 : lane = {g, n}; b[0..15] = column n, K = g*16 .. g*16+15.
//   C/D 16x16 f32: c[v] = element (M = v + 8*g, N = n).
// ---------------------------------------------------------------------------
template <int NS>
__device__ __forceinline__ void gemm_relu_layer(
    const _Float16* in, int strideIn, int KpIn,
    const _Float16* __restrict__ Wg, const float* __restrict__ bias,
    _Float16* out, int strideOut, int coutL)
{
  const int tid  = threadIdx.x;
  const int wave = tid >> 5;
  const int lane = tid & 31;
  const int m = lane & 15;     // A row within tile / B column within tile
  const int g = lane >> 4;     // K half-group selector

  constexpr int TM = NS / 16;
  const int tilesN = coutL >> 4;
  const int total  = TM * tilesN;

  for (int t = wave; t < total; t += 4) {
    const int mt = t % TM;
    const int nt = t / TM;
    v8f acc = {0.f, 0.f, 0.f, 0.f, 0.f, 0.f, 0.f, 0.f};

    for (int kt = 0; kt < KpIn; kt += 32) {
      // A fragment from LDS (two contiguous 16B chunks -> ds_load_b128)
      const _Float16* row = in + (size_t)(mt * 16 + m) * strideIn + kt;
      const v8h lo = *(const v8h*)(row + g * 16);
      const v8h hi = *(const v8h*)(row + g * 16 + 8);
      v16h a;
#pragma unroll
      for (int j = 0; j < 8; ++j) { a[j] = lo[j]; a[j + 8] = hi[j]; }
      // B fragment from global weights (contiguous 32B per lane)
      const _Float16* wc = Wg + (size_t)(nt * 16 + m) * KpIn + kt + g * 16;
      const v16h bf = *(const v16h*)wc;

      acc = __builtin_amdgcn_wmma_f32_16x16x32_f16(
          /*neg_a=*/false, a, /*neg_b=*/false, bf,
          /*c_mod=*/(short)0, acc, /*reuse_a=*/false, /*reuse_b=*/false);
    }

    const int colg = nt * 16 + m;
    const float bb = bias[colg];
#pragma unroll
    for (int v = 0; v < 8; ++v) {
      const int rowg = mt * 16 + v + 8 * g;
      float x = acc[v] + bb;
      x = x > 0.f ? x : 0.f;                       // relu
      out[(size_t)rowg * strideOut + colg] = (_Float16)x;
    }
  }
}

// ---------------------------------------------------------------------------
// Kernel 5 (x3 scales): grouped MLP + max-pool. One block per (b,s).
// X/Y are ping-pong LDS activation buffers (f16, padded row stride).
// ---------------------------------------------------------------------------
template <int NS, int C1, int C2>
__global__ __launch_bounds__(128) void mlp_maxpool_kernel(
    const float* __restrict__ xyzrgb, const float* __restrict__ new_xyz,
    const int* __restrict__ idx,
    const _Float16* __restrict__ w0, const float* __restrict__ b0,
    const _Float16* __restrict__ w1, const float* __restrict__ b1,
    const _Float16* __restrict__ w2, const float* __restrict__ b2,
    float* __restrict__ FS, int scaleOff)
{
  constexpr int KP0 = 32;                              // 6 padded to 32
  constexpr int SX  = (KP0 > C2 ? KP0 : C2) + 8;       // multiple of 8 halves
  constexpr int SY  = 128 + 8;

  __shared__ __align__(32) _Float16 Xb[NS * SX];
  __shared__ __align__(32) _Float16 Yb[NS * SY];

  const int bs  = blockIdx.x;         // b*1024 + s
  const int b   = bs >> 10;
  const int tid = threadIdx.x;

  const float qx = new_xyz[(size_t)bs * 3 + 0];
  const float qy = new_xyz[(size_t)bs * 3 + 1];
  const float qz = new_xyz[(size_t)bs * 3 + 2];

  // Build h0 rows: [rel_xyz, rgb, 0...0] in f16
  if (tid < NS) {
    const int i = idx[(size_t)bs * NS + tid];
    const float* pp = xyzrgb + ((size_t)b * Nn + i) * 6;
    _Float16* row = Xb + (size_t)tid * SX;
    row[0] = (_Float16)(pp[0] - qx);
    row[1] = (_Float16)(pp[1] - qy);
    row[2] = (_Float16)(pp[2] - qz);
    row[3] = (_Float16)pp[3];
    row[4] = (_Float16)pp[4];
    row[5] = (_Float16)pp[5];
#pragma unroll
    for (int k = 6; k < KP0; ++k) row[k] = (_Float16)0.0f;
  }
  __syncthreads();

  gemm_relu_layer<NS>(Xb, SX, KP0, w0, b0, Yb, SY, C1);
  __syncthreads();
  gemm_relu_layer<NS>(Yb, SY, C1, w1, b1, Xb, SX, C2);
  __syncthreads();
  gemm_relu_layer<NS>(Xb, SX, C2, w2, b2, Yb, SY, 128);
  __syncthreads();

  // Max-pool over the NS samples (outputs are >= 0 after relu)
  const int c = tid;                  // 0..127
  float mx = 0.0f;
  for (int r = 0; r < NS; ++r) mx = fmaxf(mx, (float)Yb[(size_t)r * SY + c]);
  FS[(size_t)bs * Cn + scaleOff + c] = mx;
}

// ---------------------------------------------------------------------------
// Kernel 6: app_feats[b,k,:] = FS[b, I1[b, I2[b,k]], :]
// ---------------------------------------------------------------------------
__global__ void app_gather_kernel(
    const float* __restrict__ FS, const int* __restrict__ I1,
    const int* __restrict__ I2, float* __restrict__ out)
{
  const int t = blockIdx.x * blockDim.x + threadIdx.x;
  if (t >= Bn * Kn * Cn) return;
  const int c   = t % Cn;
  const int rem = t / Cn;
  const int k   = rem % Kn;
  const int b   = rem / Kn;
  const int i2 = I2[b * Kn + k];
  const int i1 = I1[b * Pn + i2];
  out[t] = FS[((size_t)(b << 10) + i1) * Cn + c];
}

// ---------------------------------------------------------------------------
// Kernel 7: global_app = mean over S of FS
// ---------------------------------------------------------------------------
__global__ void global_mean_kernel(const float* __restrict__ FS,
                                   float* __restrict__ out)
{
  const int t = blockIdx.x * blockDim.x + threadIdx.x;
  if (t >= Bn * Cn) return;
  const int b = t / Cn, c = t % Cn;
  const float* p = FS + (size_t)(b << 10) * Cn + c;
  float s = 0.0f;
  for (int r = 0; r < Sn; ++r) s += p[(size_t)r * Cn];
  out[(size_t)Bn * Kn * Cn + t] = s * (1.0f / (float)Sn);
}

// ---------------------------------------------------------------------------
// Host-side launch
// ---------------------------------------------------------------------------
static inline size_t align256(size_t x) { return (x + 255) & ~(size_t)255; }

extern "C" void kernel_launch(void* const* d_in, const int* in_sizes, int n_in,
                              void* d_out, int out_size, void* d_ws, size_t ws_size,
                              hipStream_t stream)
{
  (void)in_sizes; (void)n_in; (void)out_size; (void)ws_size;

  const float* xyzrgb = (const float*)d_in[0];
  const int*   I1     = (const int*)d_in[1];
  const int*   I2     = (const int*)d_in[2];

  const float* Wf[3][3];
  const float* Bf[3][3];
  int p = 3;
  for (int s = 0; s < 3; ++s)
    for (int l = 0; l < 3; ++l) {
      Wf[s][l] = (const float*)d_in[p++];
      Bf[s][l] = (const float*)d_in[p++];
    }

  // workspace layout
  char* ws = (char*)d_ws;
  int*   fps_idx = (int*)(ws + 0);                              // 16 KB
  float* new_xyz = (float*)(ws + 16384);                        // 48 KB
  int*   idx0    = (int*)(ws + 65536);                          // 512 KB
  int*   idx1    = (int*)(ws + 65536 + 524288);                 // 1 MB
  int*   idx2    = (int*)(ws + 65536 + 524288 + 1048576);       // 2 MB
  float* FS      = (float*)(ws + 3735552);                      // 6 MB
  size_t woff    = 10027008;

  static const int couts[3][3] = {{64, 64, 128}, {64, 64, 128}, {96, 96, 128}};
  static const int cins[3][3]  = {{6, 64, 64},  {6, 64, 64},  {6, 96, 96}};
  static const int kps[3][3]   = {{32, 64, 64}, {32, 64, 64}, {32, 96, 96}};

  _Float16* Wh[3][3];
  float*    Bh[3][3];
  for (int s = 0; s < 3; ++s)
    for (int l = 0; l < 3; ++l) {
      Wh[s][l] = (_Float16*)(ws + woff);
      woff += align256((size_t)couts[s][l] * kps[s][l] * 2);
      Bh[s][l] = (float*)(ws + woff);
      woff += align256((size_t)couts[s][l] * 4);
    }

  // 1) weight repack (tiny, runs before FPS in the same stream)
  for (int s = 0; s < 3; ++s)
    for (int l = 0; l < 3; ++l) {
      const int total = couts[s][l] * kps[s][l];
      convert_weights_kernel<<<(total + 255) / 256, 256, 0, stream>>>(
          Wf[s][l], Bf[s][l], Wh[s][l], Bh[s][l],
          couts[s][l], cins[s][l], kps[s][l]);
    }

  // 2) farthest point sampling
  fps_kernel<<<Bn, FPS_T, 0, stream>>>(xyzrgb, fps_idx);

  // 3) gather sampled centers
  gather_new_xyz_kernel<<<(Bn * Sn + 127) / 128, 128, 0, stream>>>(
      xyzrgb, fps_idx, new_xyz);

  // 4) ball query, all 3 scales
  ball_query_kernel<<<dim3((Bn * Sn) / 4, 3, 1), 128, 0, stream>>>(
      xyzrgb, new_xyz, idx0, idx1, idx2);

  // 5) WMMA grouped MLP + max-pool, one kernel per scale
  mlp_maxpool_kernel<32, 64, 64><<<Bn * Sn, 128, 0, stream>>>(
      xyzrgb, new_xyz, idx0,
      Wh[0][0], Bh[0][0], Wh[0][1], Bh[0][1], Wh[0][2], Bh[0][2], FS, 0);
  mlp_maxpool_kernel<64, 64, 64><<<Bn * Sn, 128, 0, stream>>>(
      xyzrgb, new_xyz, idx1,
      Wh[1][0], Bh[1][0], Wh[1][1], Bh[1][1], Wh[1][2], Bh[1][2], FS, 128);
  mlp_maxpool_kernel<128, 96, 96><<<Bn * Sn, 128, 0, stream>>>(
      xyzrgb, new_xyz, idx2,
      Wh[2][0], Bh[2][0], Wh[2][1], Bh[2][1], Wh[2][2], Bh[2][2], FS, 256);

  // 6) outputs: app_feats then global_app
  float* out = (float*)d_out;
  app_gather_kernel<<<(Bn * Kn * Cn + 255) / 256, 256, 0, stream>>>(
      FS, I1, I2, out);
  global_mean_kernel<<<(Bn * Cn + 255) / 256, 256, 0, stream>>>(FS, out);
}
